// Tacotron_86517821211921
// MI455X (gfx1250) — matrix operand order repaired and checked by reading, not compiled, not run on hardware
//
#include <hip/hip_runtime.h>
#include <math.h>

// ---------------------------------------------------------------------------
// CDNA5 (gfx1250) wave32 WMMA helpers: D(16x16,f32) += A(16x32,f16) x B(32x16,f16)
// ---------------------------------------------------------------------------
typedef _Float16 v16h __attribute__((ext_vector_type(16)));
typedef float    v8f  __attribute__((ext_vector_type(8)));
typedef float    v4f  __attribute__((ext_vector_type(4)));

__device__ __forceinline__ v8f zero8() {
    v8f z = {0.f,0.f,0.f,0.f,0.f,0.f,0.f,0.f};
    return z;
}

// Fast, branch-free transcendentals (CDNA5 has native v_exp_f32 / v_tanh_f32 /
// v_rcp_f32 TRANS32 ops; libm versions expand to ~30 VALU + exec-mask branches).
__device__ __forceinline__ float fast_rcp(float x) {
#if __has_builtin(__builtin_amdgcn_rcpf)
    return __builtin_amdgcn_rcpf(x);
#else
    return 1.f / x;
#endif
}

__device__ __forceinline__ float sigf(float x) {
    return fast_rcp(1.f + __expf(-x));
}

__device__ __forceinline__ float tanh_fast(float x) {
#if __has_builtin(__builtin_amdgcn_tanhf)
    return __builtin_amdgcn_tanhf(x);
#else
    float t = __expf(2.f * x);
    return (t - 1.f) * fast_rcp(t + 1.f);
#endif
}

__device__ __forceinline__ v16h cvt16(v4f a0, v4f a1, v4f a2, v4f a3) {
    v16h o;
#pragma unroll
    for (int i = 0; i < 4; ++i) {
        o[i]      = (_Float16)a0[i];
        o[4 + i]  = (_Float16)a1[i];
        o[8 + i]  = (_Float16)a2[i];
        o[12 + i] = (_Float16)a3[i];
    }
    return o;
}

// FAST path: K % 32 == 0, full 16-row A tile and 16-col B tile in bounds.
// Unconditional 128-bit loads (16B-aligned: all row strides are multiples of 4 floats).
// A-frag: lane = row (lane&15), half = lane>>4; elems 0..7 -> K=k0+8h+e,
//         elems 8..15 -> K=k0+16+8h+e. B-frag: lane = col; same K mapping as A.
__device__ __forceinline__ v8f wmma_tile_fast(const float* __restrict__ X, int ldx, int m0,
                                              const float* __restrict__ W, int ldw, int n0,
                                              int K, v8f acc) {
    const int lane = threadIdx.x & 31;
    const int r = lane & 15;
    const int h = lane >> 4;
    const float* xr = X + (size_t)(m0 + r) * ldx;
    const float* wr = W + (size_t)(n0 + r) * ldw;
    for (int k0 = 0; k0 < K; k0 += 32) {
        const float* xa = xr + k0 + 8 * h;
        const float* xb = xr + k0 + 16 + 8 * h;
        const float* wb = wr + k0 + 8 * h;
        const float* wc = wr + k0 + 16 + 8 * h;
        v4f a0 = *(const v4f*)(xa);
        v4f a1 = *(const v4f*)(xa + 4);
        v4f a2 = *(const v4f*)(xb);
        v4f a3 = *(const v4f*)(xb + 4);
        v4f b0 = *(const v4f*)(wb);
        v4f b1 = *(const v4f*)(wb + 4);
        v4f b2 = *(const v4f*)(wc);
        v4f b3 = *(const v4f*)(wc + 4);
        v16h a = cvt16(a0, a1, a2, a3);
        v16h b = cvt16(b0, b1, b2, b3);
        acc = __builtin_amdgcn_wmma_f32_16x16x32_f16(false, a, false, b, (short)0, acc,
                                                     false, false);
    }
    return acc;
}

// SAFE path (K or M/N tails): clamp addresses (always in-bounds load), zero the
// value with a post-load select -> v_cndmask, no divergent branches.
__device__ __forceinline__ v8f wmma_tile_safe(const float* __restrict__ X, int ldx, int m0, int M,
                                              const float* __restrict__ W, int ldw, int n0, int N,
                                              int K, v8f acc) {
    const int lane = threadIdx.x & 31;
    const int r = lane & 15;
    const int h = lane >> 4;
    const int am = m0 + r;
    const int bn = n0 + r;
    const bool mv = am < M;
    const bool nv = bn < N;
    const float* xr = X + (size_t)(mv ? am : (M - 1)) * ldx;
    const float* wr = W + (size_t)(nv ? bn : (N - 1)) * ldw;
    for (int k0 = 0; k0 < K; k0 += 32) {
        v16h a, b;
#pragma unroll
        for (int e = 0; e < 8; ++e) {
            int ka0 = k0 + 8 * h + e;
            int ka1 = k0 + 16 + 8 * h + e;
            int kb0 = k0 + 8 * h + e;
            int kb1 = k0 + 16 + 8 * h + e;
            float xa0 = xr[ka0 < K ? ka0 : 0];
            float xa1 = xr[ka1 < K ? ka1 : 0];
            float wb0 = wr[kb0 < K ? kb0 : 0];
            float wb1 = wr[kb1 < K ? kb1 : 0];
            a[e]     = (_Float16)((mv && ka0 < K) ? xa0 : 0.f);
            a[e + 8] = (_Float16)((mv && ka1 < K) ? xa1 : 0.f);
            b[e]     = (_Float16)((nv && kb0 < K) ? wb0 : 0.f);
            b[e + 8] = (_Float16)((nv && kb1 < K) ? wb1 : 0.f);
        }
        acc = __builtin_amdgcn_wmma_f32_16x16x32_f16(false, a, false, b, (short)0, acc,
                                                     false, false);
    }
    return acc;
}

// C/D layout: lane holds col n0 + (lane&15); VGPR v holds row m0 + v + 8*(lane>>4)
__device__ __forceinline__ void wmma_store_fast(float* __restrict__ Y, int ldy, int m0, int n0,
                                                v8f acc, const float* b1, const float* b2,
                                                int act) {
    const int lane = threadIdx.x & 31;
    const int c = n0 + (lane & 15);
    const int h = lane >> 4;
    float bias = 0.f;
    if (b1) bias += b1[c];
    if (b2) bias += b2[c];
#pragma unroll
    for (int v = 0; v < 8; ++v) {
        int rm = m0 + v + 8 * h;
        float val = acc[v] + bias;
        if (act == 1)      val = fmaxf(val, 0.f);
        else if (act == 2) val = sigf(val);
        else if (act == 3) val = tanh_fast(val);
        Y[(size_t)rm * ldy + c] = val;
    }
}

__device__ __forceinline__ void wmma_store_safe(float* __restrict__ Y, int ldy, int m0, int M,
                                                int n0, int N, v8f acc,
                                                const float* b1, const float* b2, int act) {
    const int lane = threadIdx.x & 31;
    const int c = n0 + (lane & 15);
    const int h = lane >> 4;
    if (c >= N) return;
    float bias = 0.f;
    if (b1) bias += b1[c];
    if (b2) bias += b2[c];
#pragma unroll
    for (int v = 0; v < 8; ++v) {
        int rm = m0 + v + 8 * h;
        if (rm < M) {
            float val = acc[v] + bias;
            if (act == 1)      val = fmaxf(val, 0.f);
            else if (act == 2) val = sigf(val);
            else if (act == 3) val = tanh_fast(val);
            Y[(size_t)rm * ldy + c] = val;
        }
    }
}

// ---------------------------------------------------------------------------
// Generic GEMM: Y[M,N] = act(X[M,K] @ W[N,K]^T + bias). One wave per 16x16 tile.
// ---------------------------------------------------------------------------
__global__ __launch_bounds__(256) void k_gemm(const float* __restrict__ X, int ldx,
                                              const float* __restrict__ W, int ldw,
                                              const float* __restrict__ bias,
                                              float* __restrict__ Y, int ldy,
                                              int M, int N, int K, int act) {
    int wave = blockIdx.x * (blockDim.x >> 5) + (threadIdx.x >> 5);
    int tn = (N + 15) >> 4;
    int tm = (M + 15) >> 4;
    if (wave >= tn * tm) return;       // wave-uniform
    int ti = wave / tn, tj = wave % tn;
    v8f acc = zero8();
    bool aligned = ((K & 31) == 0) && (ti * 16 + 16 <= M) && (tj * 16 + 16 <= N);
    if (aligned) {
        acc = wmma_tile_fast(X, ldx, ti * 16, W, ldw, tj * 16, K, acc);
        wmma_store_fast(Y, ldy, ti * 16, tj * 16, acc, bias, nullptr, act);
    } else {
        acc = wmma_tile_safe(X, ldx, ti * 16, M, W, ldw, tj * 16, N, K, acc);
        wmma_store_safe(Y, ldy, ti * 16, M, tj * 16, N, acc, bias, nullptr, act);
    }
}

// ---------------------------------------------------------------------------
// Elementwise / conv kernels (tensors stored [B*T, C] row-major unless noted)
// ---------------------------------------------------------------------------
__global__ void k_embed(const int* __restrict__ x, const float* __restrict__ emb,
                        const float* __restrict__ se, float* __restrict__ y,
                        int B, int T, int E) {
    int i = blockIdx.x * blockDim.x + threadIdx.x;
    int tot = B * T * 2 * E;
    if (i >= tot) return;
    int c = i % (2 * E);
    int bt = i / (2 * E);
    int b = bt / T;
    float v;
    if (c < E) {
        int id = x[bt];
        if (id < 0) id = 0;
        v = emb[(size_t)id * E + c];
    } else {
        v = se[(size_t)b * E + (c - E)];
    }
    y[i] = v;
}

// conv bank (kernel k, pad k/2, crop to T) + ReLU + BN(eval). x:[B*T,cin]
__global__ void k_conv_bank(const float* __restrict__ x, const float* __restrict__ w,
                            const float* __restrict__ g, const float* __restrict__ be,
                            float* __restrict__ y, int B, int T, int cin, int ch,
                            int k, int ldy, int coff) {
    int i = blockIdx.x * blockDim.x + threadIdx.x;
    if (i >= B * T * ch) return;
    int o = i % ch;
    int bt = i / ch;
    int t = bt % T;
    int pad = k / 2;
    float s = 0.f;
    for (int j = 0; j < k; ++j) {
        int tt = t - pad + j;
        if (tt < 0 || tt >= T) continue;
        const float* xr = x + (size_t)(bt - t + tt) * cin;
        const float* wr = w + ((size_t)o * cin) * k + j;
        for (int ic = 0; ic < cin; ++ic) s += xr[ic] * wr[(size_t)ic * k];
    }
    s = fmaxf(s, 0.f);
    float bscale = rsqrtf(1.f + 1e-5f);
    y[(size_t)bt * ldy + coff + o] = s * (g[o] * bscale) + be[o];
}

// maxpool k=2 stride=1 pad=1 (crop T): out[t] = max(x[t-1], x[t])
__global__ void k_maxpool(const float* __restrict__ x, float* __restrict__ y,
                          int B, int T, int C) {
    int i = blockIdx.x * blockDim.x + threadIdx.x;
    if (i >= B * T * C) return;
    int c = i % C;
    int bt = i / C;
    int t = bt % T;
    float v = x[(size_t)bt * C + c];
    if (t > 0) v = fmaxf(v, x[(size_t)(bt - 1) * C + c]);
    y[i] = v;
}

// k=3 conv + (optional ReLU) + BN + (optional residual)
__global__ void k_conv3(const float* __restrict__ x, int cin, const float* __restrict__ w,
                        const float* __restrict__ g, const float* __restrict__ be,
                        const float* __restrict__ res, float* __restrict__ y,
                        int B, int T, int cout, int relu_) {
    int i = blockIdx.x * blockDim.x + threadIdx.x;
    if (i >= B * T * cout) return;
    int o = i % cout;
    int bt = i / cout;
    int t = bt % T;
    float s = 0.f;
    for (int j = 0; j < 3; ++j) {
        int tt = t - 1 + j;
        if (tt < 0 || tt >= T) continue;
        const float* xr = x + (size_t)(bt - t + tt) * cin;
        const float* wr = w + ((size_t)o * cin) * 3 + j;
        for (int ic = 0; ic < cin; ++ic) s += xr[ic] * wr[(size_t)ic * 3];
    }
    if (relu_) s = fmaxf(s, 0.f);
    float bscale = rsqrtf(1.f + 1e-5f);
    s = s * (g[o] * bscale) + be[o];
    if (res) s += res[(size_t)bt * cout + o];
    y[(size_t)bt * cout + o] = s;
}

__global__ void k_copy(const float* __restrict__ x, float* __restrict__ y, int n) {
    int i = blockIdx.x * blockDim.x + threadIdx.x;
    if (i < n) y[i] = x[i];
}

// highway combine (in place): y = sig(h2)*relu(h1) + (1-sig(h2))*y
__global__ void k_hwy(float* __restrict__ y, const float* __restrict__ h1,
                      const float* __restrict__ h2, int n) {
    int i = blockIdx.x * blockDim.x + threadIdx.x;
    if (i >= n) return;
    float gg = sigf(h2[i]);
    y[i] = gg * fmaxf(h1[i], 0.f) + (1.f - gg) * y[i];
}

// [B,C,T] -> [B*T,C]
__global__ void k_bct_to_btc(const float* __restrict__ x, float* __restrict__ y,
                             int B, int C, int T) {
    int i = blockIdx.x * blockDim.x + threadIdx.x;
    if (i >= B * T * C) return;
    int c = i % C;
    int bt = i / C;
    int t = bt % T;
    int b = bt / T;
    y[i] = x[((size_t)b * C + c) * T + t];
}

// [B*T,F] -> [B,F,T]
__global__ void k_btc_to_bct(const float* __restrict__ x, float* __restrict__ y,
                             int B, int F, int T) {
    int i = blockIdx.x * blockDim.x + threadIdx.x;
    if (i >= B * T * F) return;
    int f = i % F;
    int bt = i / F;
    int t = bt % T;
    int b = bt / T;
    y[((size_t)b * F + f) * T + t] = x[i];
}

// ---------------------------------------------------------------------------
// Persistent bi-GRU direction kernel (single workgroup). H <= 128, B = 16.
// gi = X@Wih^T + b_ih precomputed ([B*T,3H]); recurrent gh done with WMMA from LDS.
// ---------------------------------------------------------------------------
__global__ __launch_bounds__(256) void k_gru_seq(const float* __restrict__ gi,
                                                 const float* __restrict__ w_hh,
                                                 const float* __restrict__ b_hh,
                                                 float* __restrict__ out, int ldo, int ooff,
                                                 int B, int T, int H, int rev) {
    __shared__ __align__(16) float hs[16 * 128];
    __shared__ __align__(16) float gh[16 * 384];
    const int tid = threadIdx.x, nt = blockDim.x;
    const int wv = tid >> 5, nw = nt >> 5;
    for (int i = tid; i < B * H; i += nt) hs[i] = 0.f;
    __syncthreads();
    const int tn = (3 * H + 15) >> 4;
    for (int t = 0; t < T; ++t) {
        int ta = rev ? (T - 1 - t) : t;
        for (int tj = wv; tj < tn; tj += nw) {
            v8f acc = zero8();
            acc = wmma_tile_fast(hs, H, 0, w_hh, H, tj * 16, H, acc);
            wmma_store_fast(gh, 3 * H, 0, tj * 16, acc, b_hh, nullptr, 0);
        }
        __syncthreads();
        for (int i = tid; i < B * H; i += nt) {
            int b = i / H, j = i % H;
            const float* gir = gi + (size_t)(b * T + ta) * (3 * H);
            const float* ghr = gh + (size_t)b * (3 * H);
            float r = sigf(gir[j] + ghr[j]);
            float z = sigf(gir[H + j] + ghr[H + j]);
            float n = tanh_fast(gir[2 * H + j] + r * ghr[2 * H + j]);
            float hv = (1.f - z) * n + z * hs[i];
            hs[i] = hv;
            out[(size_t)(b * T + ta) * ldo + ooff + j] = hv;
        }
        __syncthreads();
    }
}

// ---------------------------------------------------------------------------
// Persistent decoder kernel: 200 sequential steps, one workgroup of 512.
// All recurrent state in global scratch (L2-resident), phase-barriered.
// ---------------------------------------------------------------------------
struct DecArgs {
    const float* m;      // [B,80,STEPS]
    const float* enc;    // [B*T,256]
    const float* encp;   // [B*T,256]
    const float *p_fc1w, *p_fc1b, *p_fc2w, *p_fc2b;                // prenet
    const float *convw, *Lw, *Lb, *Ww, *Wb, *vw;                   // LSA
    const float *ag_wih, *ag_whh, *ag_bih, *ag_bhh;                // attn GRU
    const float *ri_w, *ri_b;                                      // rnn_input
    const float *l1_wih, *l1_whh, *l1_bih, *l1_bhh;                // lstm1
    const float *l2_wih, *l2_whh, *l2_bih, *l2_bhh;                // lstm2
    const float* melw;                                             // [1600,512]
    float* mel_out;      // [B,80,STEPS]
    float* attn_out;     // [B,N,T]
    float* S;            // scratch
    int T, steps, r, N;
};

__global__ __launch_bounds__(512) void k_decoder(DecArgs A) {
    const int T = A.T;
    float* attn_h = A.S;                // [16,256]
    float* ctx    = attn_h + 16 * 256;  // [16,256]
    float* cum    = ctx + 16 * 256;     // [16,T]
    float* att    = cum + 16 * T;       // [16,T]
    float* r1h    = att + 16 * T;       // [16,512]
    float* r1c    = r1h + 16 * 512;
    float* r2h    = r1c + 16 * 512;
    float* r2c    = r2h + 16 * 512;
    float* xbuf   = r2c + 16 * 512;     // [16,512] concat staging
    float* xx     = xbuf + 16 * 512;    // [16,512]
    float* gates  = xx + 16 * 512;      // [16,2048]
    float* pq     = gates + 16 * 2048;  // [16,256]
    float* preb   = pq + 16 * 256;      // [16,80]
    float* ub     = preb + 16 * 96;     // [16,T]
    float* h1     = ub + 16 * T;        // [16,256]
    __shared__ float ssum[16];
    const int tid = threadIdx.x, nt = blockDim.x, wv = tid >> 5;

    for (int i = tid; i < 16 * 256; i += nt) { attn_h[i] = 0.f; ctx[i] = 0.f; }
    for (int i = tid; i < 16 * T; i += nt)   { cum[i] = 0.f; att[i] = 0.f; }
    for (int i = tid; i < 16 * 512; i += nt) { r1h[i] = 0.f; r1c[i] = 0.f; r2h[i] = 0.f; r2c[i] = 0.f; }
    __syncthreads();

    for (int s = 0; s < A.N; ++s) {
        // -------- prenet input (go frame or m[:, :, r*s-1]) --------
        for (int i = tid; i < 16 * 80; i += nt) {
            int b = i / 80, c = i % 80;
            preb[b * 80 + c] = (s == 0) ? 0.f
                : A.m[((size_t)b * 80 + c) * A.steps + (A.r * s - 1)];
            // hint the next step's frame into cache (global_prefetch_b8)
            if (s + 1 < A.N)
                __builtin_prefetch(&A.m[((size_t)b * 80 + c) * A.steps + (A.r * (s + 1) - 1)], 0, 1);
        }
        __syncthreads();
        { // fc1: [16,80] -> [16,256], relu  (K=80 tail -> safe path)
            v8f acc = zero8();
            acc = wmma_tile_safe(preb, 80, 0, 16, A.p_fc1w, 80, wv * 16, 256, 80, acc);
            wmma_store_fast(h1, 256, 0, wv * 16, acc, A.p_fc1b, nullptr, 1);
        }
        __syncthreads();
        if (wv < 8) { // fc2 -> xbuf cols [256,384), relu
            v8f acc = zero8();
            acc = wmma_tile_fast(h1, 256, 0, A.p_fc2w, 256, wv * 16, 256, acc);
            wmma_store_fast(xbuf + 256, 512, 0, wv * 16, acc, A.p_fc2b, nullptr, 1);
        }
        // OLD ctx into xbuf cols [0,256)
        for (int i = tid; i < 16 * 256; i += nt) {
            int b = i / 256, c = i % 256;
            xbuf[b * 512 + c] = ctx[i];
        }
        __syncthreads();
        // -------- attention GRU: gi (48 tiles) + gh (48 tiles) --------
        for (int q = wv; q < 96; q += 16) {
            v8f acc = zero8();
            if (q < 48) {
                acc = wmma_tile_fast(xbuf, 512, 0, A.ag_wih, 384, q * 16, 384, acc);
                wmma_store_fast(gates, 2048, 0, q * 16, acc, A.ag_bih, nullptr, 0);
            } else {
                int j = q - 48;
                acc = wmma_tile_fast(attn_h, 256, 0, A.ag_whh, 256, j * 16, 256, acc);
                wmma_store_fast(gates + 768, 2048, 0, j * 16, acc, A.ag_bhh, nullptr, 0);
            }
        }
        __syncthreads();
        for (int i = tid; i < 16 * 256; i += nt) {
            int b = i / 256, j = i % 256;
            const float* gi = gates + (size_t)b * 2048;
            const float* gh = gi + 768;
            float r = sigf(gi[j] + gh[j]);
            float z = sigf(gi[256 + j] + gh[256 + j]);
            float n = tanh_fast(gi[512 + j] + r * gh[512 + j]);
            attn_h[i] = (1.f - z) * n + z * attn_h[i];
        }
        __syncthreads();
        { // pq = attn_h @ Ww^T + Wb
            v8f acc = zero8();
            acc = wmma_tile_fast(attn_h, 256, 0, A.Ww, 256, wv * 16, 256, acc);
            wmma_store_fast(pq, 256, 0, wv * 16, acc, A.Wb, nullptr, 0);
        }
        __syncthreads();
        // -------- location-sensitive attention energies --------
        for (int i = tid; i < 16 * T; i += nt) {
            int b = i / T, t = i % T;
            float cl[32];
            for (int c = 0; c < 32; ++c) {
                float sgm = 0.f;
                for (int j = 0; j < 31; ++j) {
                    int tt = t - 15 + j;
                    if (tt < 0 || tt >= T) continue;
                    sgm += cum[b * T + tt] * A.convw[(c * 2 + 0) * 31 + j]
                         + att[b * T + tt] * A.convw[(c * 2 + 1) * 31 + j];
                }
                cl[c] = sgm;
            }
            const float* ep = A.encp + ((size_t)(b * T + t)) * 256;
            float accu = 0.f;
            for (int d = 0; d < 256; ++d) {
                float s2 = pq[b * 256 + d] + ep[d] + A.Lb[d];
                const float* lw = A.Lw + (size_t)d * 32;
                for (int c = 0; c < 32; ++c) s2 += lw[c] * cl[c];
                accu += A.vw[d] * tanh_fast(s2);
            }
            ub[i] = sigf(accu);
        }
        __syncthreads();
        if (tid < 16) {
            float sm = 0.f;
            for (int t = 0; t < T; ++t) sm += ub[tid * T + t];
            ssum[tid] = sm;
        }
        __syncthreads();
        for (int i = tid; i < 16 * T; i += nt) {
            int b = i / T, t = i % T;
            float sc = ub[i] / ssum[b];
            att[i] = sc;
            cum[i] += sc;
            A.attn_out[((size_t)b * A.N + s) * T + t] = sc;
        }
        __syncthreads();
        // ctx = scores @ enc
        for (int i = tid; i < 16 * 256; i += nt) {
            int b = i / 256, d = i % 256;
            float sm = 0.f;
            const float* er = A.enc + (size_t)b * T * 256 + d;
            for (int t = 0; t < T; ++t) sm += att[b * T + t] * er[(size_t)t * 256];
            ctx[i] = sm;
        }
        __syncthreads();
        // concat(new ctx, attn_h) for rnn_input
        for (int i = tid; i < 16 * 256; i += nt) {
            int b = i / 256, c = i % 256;
            xbuf[b * 512 + c] = ctx[i];
            xbuf[b * 512 + 256 + c] = attn_h[i];
        }
        __syncthreads();
        for (int q = wv; q < 32; q += 16) { // xx = rnn_input
            v8f acc = zero8();
            acc = wmma_tile_fast(xbuf, 512, 0, A.ri_w, 512, q * 16, 512, acc);
            wmma_store_fast(xx, 512, 0, q * 16, acc, A.ri_b, nullptr, 0);
        }
        __syncthreads();
        // -------- LSTM1 (two accumulated WMMA passes) --------
        for (int q = wv; q < 128; q += 16) {
            v8f acc = zero8();
            acc = wmma_tile_fast(xx, 512, 0, A.l1_wih, 512, q * 16, 512, acc);
            acc = wmma_tile_fast(r1h, 512, 0, A.l1_whh, 512, q * 16, 512, acc);
            wmma_store_fast(gates, 2048, 0, q * 16, acc, A.l1_bih, A.l1_bhh, 0);
        }
        __syncthreads();
        for (int i = tid; i < 16 * 512; i += nt) {
            int b = i / 512, j = i % 512;
            const float* g = gates + (size_t)b * 2048;
            float cc = sigf(g[512 + j]) * r1c[i] + sigf(g[j]) * tanh_fast(g[1024 + j]);
            float hh = sigf(g[1536 + j]) * tanh_fast(cc);
            r1c[i] = cc; r1h[i] = hh; xx[i] += hh;
        }
        __syncthreads();
        // -------- LSTM2 --------
        for (int q = wv; q < 128; q += 16) {
            v8f acc = zero8();
            acc = wmma_tile_fast(xx, 512, 0, A.l2_wih, 512, q * 16, 512, acc);
            acc = wmma_tile_fast(r2h, 512, 0, A.l2_whh, 512, q * 16, 512, acc);
            wmma_store_fast(gates, 2048, 0, q * 16, acc, A.l2_bih, A.l2_bhh, 0);
        }
        __syncthreads();
        for (int i = tid; i < 16 * 512; i += nt) {
            int b = i / 512, j = i % 512;
            const float* g = gates + (size_t)b * 2048;
            float cc = sigf(g[512 + j]) * r2c[i] + sigf(g[j]) * tanh_fast(g[1024 + j]);
            float hh = sigf(g[1536 + j]) * tanh_fast(cc);
            r2c[i] = cc; r2h[i] = hh; xx[i] += hh;
        }
        __syncthreads();
        // -------- mel projection with fused reshape(B,80,20)[:, :, :r] store --------
        for (int q = wv; q < 100; q += 16) {
            v8f acc = zero8();
            acc = wmma_tile_fast(xx, 512, 0, A.melw, 512, q * 16, 512, acc);
            int lane = tid & 31;
            int c = q * 16 + (lane & 15);
            int hh = lane >> 4;
            int nm = c / 20, jj = c % 20;
            if (jj < A.r) {
#pragma unroll
                for (int v = 0; v < 8; ++v) {
                    int b = v + 8 * hh;
                    A.mel_out[((size_t)b * 80 + nm) * A.steps + s * A.r + jj] = acc[v];
                }
            }
        }
        __syncthreads();
    }
}

// ---------------------------------------------------------------------------
// Host-side orchestration
// ---------------------------------------------------------------------------
static inline void launch_gemm(hipStream_t st, const float* X, int ldx, const float* W,
                               int ldw, const float* b, float* Y, int ldy,
                               int M, int N, int K, int act) {
    int waves = ((M + 15) / 16) * ((N + 15) / 16);
    int blocks = (waves + 7) / 8;
    k_gemm<<<blocks, 256, 0, st>>>(X, ldx, W, ldw, b, Y, ldy, M, N, K, act);
}

static void run_cbhg(hipStream_t st, void* const* din, int base, int K, int cin, int ch,
                     int p1, int p2, int nhw, bool prehw, const float* x, int B, int T,
                     float* out, char* ws, size_t& off) {
    auto F = [&](int i) { return (const float*)din[i]; };
    auto carve = [&](size_t n) {
        float* p = (float*)(ws + off);
        off += (n * sizeof(float) + 255) & ~(size_t)255;
        return p;
    };
    const int BT = B * T;
    float* ycat = carve((size_t)BT * K * ch);
    float* pool = carve((size_t)BT * K * ch);
    float* pr1  = carve((size_t)BT * p1);
    float* pr2  = carve((size_t)BT * p2);
    float* hwy  = carve((size_t)BT * ch);
    float* hh1  = carve((size_t)BT * ch);
    float* hh2  = carve((size_t)BT * ch);
    float* gif  = carve((size_t)BT * 3 * ch);
    float* gib  = carve((size_t)BT * 3 * ch);

    int nb = (BT * ch + 255) / 256;
    for (int k = 1; k <= K; ++k) {
        k_conv_bank<<<nb, 256, 0, st>>>(x, F(base + 3 * (k - 1)), F(base + 3 * (k - 1) + 1),
                                        F(base + 3 * (k - 1) + 2), ycat, B, T, cin, ch, k,
                                        K * ch, (k - 1) * ch);
    }
    int nkc = (BT * K * ch + 255) / 256;
    k_maxpool<<<nkc, 256, 0, st>>>(ycat, pool, B, T, K * ch);
    int pbase = base + 3 * K;
    int n1 = (BT * p1 + 255) / 256;
    k_conv3<<<n1, 256, 0, st>>>(pool, K * ch, F(pbase), F(pbase + 1), F(pbase + 2),
                                nullptr, pr1, B, T, p1, 1);
    int n2 = (BT * p2 + 255) / 256;
    k_conv3<<<n2, 256, 0, st>>>(pr1, p1, F(pbase + 3), F(pbase + 4), F(pbase + 5),
                                x, pr2, B, T, p2, 0);
    int hb = base + 3 * K + 6;
    int gf = hb + 4 * nhw;
    if (prehw) {
        launch_gemm(st, pr2, p2, F(gf + 8), p2, nullptr, hwy, ch, BT, ch, p2, 0);
    } else {
        k_copy<<<nb, 256, 0, st>>>(pr2, hwy, BT * ch);
    }
    for (int j = 0; j < nhw; ++j) {
        launch_gemm(st, hwy, ch, F(hb + 4 * j + 0), ch, F(hb + 4 * j + 1), hh1, ch, BT, ch, ch, 0);
        launch_gemm(st, hwy, ch, F(hb + 4 * j + 2), ch, F(hb + 4 * j + 3), hh2, ch, BT, ch, ch, 0);
        k_hwy<<<nb, 256, 0, st>>>(hwy, hh1, hh2, BT * ch);
    }
    // bi-GRU: precompute input projections, then persistent recurrent kernels
    launch_gemm(st, hwy, ch, F(gf + 0), ch, F(gf + 2), gif, 3 * ch, BT, 3 * ch, ch, 0);
    launch_gemm(st, hwy, ch, F(gf + 4), ch, F(gf + 6), gib, 3 * ch, BT, 3 * ch, ch, 0);
    k_gru_seq<<<1, 256, 0, st>>>(gif, F(gf + 1), F(gf + 3), out, 2 * ch, 0, B, T, ch, 0);
    k_gru_seq<<<1, 256, 0, st>>>(gib, F(gf + 5), F(gf + 7), out, 2 * ch, ch, B, T, ch, 1);
}

extern "C" void kernel_launch(void* const* d_in, const int* in_sizes, int n_in,
                              void* d_out, int out_size, void* d_ws, size_t ws_size,
                              hipStream_t stream) {
    (void)in_sizes; (void)n_in; (void)out_size; (void)ws_size;
    const int B = 16, TE = 128, STEPS = 400, R = 2, NSTEP = STEPS / R, E = 256, NM = 80;
    auto F = [&](int i) { return (const float*)d_in[i]; };
    const int* xi = (const int*)d_in[0];
    const float* m  = F(1);
    const float* se = F(2);

    char* ws = (char*)d_ws;
    size_t off = 0;
    auto carve = [&](size_t n) {
        float* p = (float*)(ws + off);
        off += (n * sizeof(float) + 255) & ~(size_t)255;
        return p;
    };
    float* ecat = carve((size_t)B * TE * 512);
    float* eh1  = carve((size_t)B * TE * 256);
    float* eh   = carve((size_t)B * TE * 128);
    float* enc  = carve((size_t)B * TE * 256);
    float* encp = carve((size_t)B * TE * 256);
    // decoder scratch: states + gates + staging
    size_t dec_elems = 16 * 256 * 2 + 16 * TE * 2 + 16 * 512 * 6 + 16 * 2048
                     + 16 * 256 + 16 * 96 + 16 * TE + 16 * 256;
    float* decS     = carve(dec_elems);
    float* mel_btc  = carve((size_t)B * STEPS * NM);
    float* post     = carve((size_t)B * STEPS * 256);
    float* lin      = carve((size_t)B * STEPS * 513);

    float* out_mel  = (float*)d_out;                        // [B,80,400]
    float* out_lin  = out_mel + (size_t)B * NM * STEPS;     // [B,513,400]
    float* out_attn = out_lin + (size_t)B * 513 * STEPS;    // [B,N,128]

    // ---------------- encoder ----------------
    {
        int n = B * TE * 2 * E;
        k_embed<<<(n + 255) / 256, 256, 0, stream>>>(xi, F(3), se, ecat, B, TE, E);
    }
    launch_gemm(stream, ecat, 512, F(4), 512, F(5), eh1, 256, B * TE, 256, 512, 1);
    launch_gemm(stream, eh1, 256, F(6), 256, F(7), eh, 128, B * TE, 128, 256, 1);
    size_t mark = off;
    run_cbhg(stream, d_in, 8, 16, 128, 128, 128, 128, 4, false, eh, B, TE, enc, ws, off);
    launch_gemm(stream, enc, 256, F(86), 256, nullptr, encp, 256, B * TE, 256, 256, 0);

    // ---------------- decoder ----------------
    DecArgs A;
    A.m = m; A.enc = enc; A.encp = encp;
    A.p_fc1w = F(87); A.p_fc1b = F(88); A.p_fc2w = F(89); A.p_fc2b = F(90);
    A.convw = F(91); A.Lw = F(92); A.Lb = F(93); A.Ww = F(94); A.Wb = F(95); A.vw = F(96);
    A.ag_wih = F(97); A.ag_whh = F(98); A.ag_bih = F(99); A.ag_bhh = F(100);
    A.ri_w = F(101); A.ri_b = F(102);
    A.l1_wih = F(103); A.l1_whh = F(104); A.l1_bih = F(105); A.l1_bhh = F(106);
    A.l2_wih = F(107); A.l2_whh = F(108); A.l2_bih = F(109); A.l2_bhh = F(110);
    A.melw = F(111);
    A.mel_out = out_mel; A.attn_out = out_attn; A.S = decS;
    A.T = TE; A.steps = STEPS; A.r = R; A.N = NSTEP;
    k_decoder<<<1, 512, 0, stream>>>(A);

    // ---------------- postnet ----------------
    {
        int n = B * STEPS * NM;
        k_bct_to_btc<<<(n + 255) / 256, 256, 0, stream>>>(out_mel, mel_btc, B, NM, STEPS);
    }
    off = mark;  // reuse CBHG scratch region
    run_cbhg(stream, d_in, 112, 8, 80, 128, 256, 80, 4, true, mel_btc, B, STEPS, post, ws, off);
    launch_gemm(stream, post, 256, F(167), 256, nullptr, lin, 513, B * STEPS, 513, 256, 0);
    {
        int n = B * STEPS * 513;
        k_btc_to_bct<<<(n + 255) / 256, 256, 0, stream>>>(lin, out_lin, B, 513, STEPS);
    }
}
